// MoELayer_65515431133732
// MI455X (gfx1250) — compile-verified, matrix operand order checked by reading, not run on hardware
//
#include <hip/hip_runtime.h>
#include <hip/hip_bf16.h>
#include <math.h>

// Problem constants (match reference)
static constexpr int NTOK = 8192;   // B*S
static constexpr int DIM  = 1024;   // D
static constexpr int NEXP = 8;      // E
static constexpr int HID  = 4096;   // H
static constexpr int DOUT = 1024;   // DO
static constexpr float LN_EPS = 1e-5f;

// Workspace layout (bytes)
static constexpr size_t OFF_XB  = 0;                                        // x bf16 [NTOK][DIM]
static constexpr size_t OFF_W1T = OFF_XB  + (size_t)NTOK * DIM * 2;         // W1^T bf16 [E][H][D]
static constexpr size_t OFF_W2T = OFF_W1T + (size_t)NEXP * HID * DIM * 2;   // W2^T bf16 [E][DO][H]
static constexpr size_t OFF_WST = OFF_W2T + (size_t)NEXP * DOUT * HID * 2;  // Ws^T bf16 [DO][D]
static constexpr size_t OFF_H   = OFF_WST + (size_t)DOUT * DIM * 2;         // h bf16 [NTOK][HID] (per-expert reuse)
static constexpr size_t OFF_IDX = OFF_H   + (size_t)NTOK * HID * 2;         // int  [E][NTOK]
static constexpr size_t OFF_WGT = OFF_IDX + (size_t)NEXP * NTOK * 4;        // f32  [E][NTOK]
static constexpr size_t OFF_CNT = OFF_WGT + (size_t)NEXP * NTOK * 4;        // int  [E]

typedef __attribute__((ext_vector_type(16))) __bf16 bf16x16;
typedef __attribute__((ext_vector_type(8)))  __bf16 bf16x8;
typedef __attribute__((ext_vector_type(8)))  float  f32x8;

__device__ __forceinline__ float gelu_exact(float x) {
    return 0.5f * x * (1.0f + erff(x * 0.70710678118654752f));
}

// Load one 16-element bf16 fragment: two contiguous 16B groups (k .. k+7, k+16 .. k+23)
__device__ __forceinline__ bf16x16 load_frag(const __bf16* p) {
    bf16x8 lo = *reinterpret_cast<const bf16x8*>(p);
    bf16x8 hi = *reinterpret_cast<const bf16x8*>(p + 16);
    return __builtin_shufflevector(lo, hi, 0,1,2,3,4,5,6,7,8,9,10,11,12,13,14,15);
}

// ---------------- small utility kernels ----------------

__global__ __launch_bounds__(32) void k_init_counts(int* counts) {
    if (threadIdx.x < NEXP) counts[threadIdx.x] = 0;
}

__global__ __launch_bounds__(256) void k_convert_x(const float* __restrict__ x,
                                                   __bf16* __restrict__ xb) {
    size_t i = ((size_t)blockIdx.x * 256 + threadIdx.x) * 4;
    if (i + 3 < (size_t)NTOK * DIM) {
        float4 v = *reinterpret_cast<const float4*>(x + i);
        xb[i + 0] = (__bf16)v.x;
        xb[i + 1] = (__bf16)v.y;
        xb[i + 2] = (__bf16)v.z;
        xb[i + 3] = (__bf16)v.w;
    }
}

// in  f32 [Bt][R][C]  ->  out bf16 [Bt][C][R]   (LDS-tiled 32x32 transpose)
__global__ __launch_bounds__(256) void k_transpose_bf16(const float* __restrict__ in,
                                                        __bf16* __restrict__ out,
                                                        int R, int C) {
    __shared__ float tile[32][33];
    const size_t bofs = (size_t)blockIdx.z * R * C;
    int r0 = blockIdx.y * 32, c0 = blockIdx.x * 32;
    #pragma unroll
    for (int i = 0; i < 4; ++i) {
        int r = r0 + threadIdx.y + i * 8;
        tile[threadIdx.y + i * 8][threadIdx.x] = in[bofs + (size_t)r * C + c0 + threadIdx.x];
    }
    __syncthreads();
    #pragma unroll
    for (int i = 0; i < 4; ++i) {
        int c = c0 + threadIdx.y + i * 8;
        out[bofs + (size_t)c * R + r0 + threadIdx.x] = (__bf16)tile[threadIdx.x][threadIdx.y + i * 8];
    }
}

// ---------------- gate: logits -> softmax -> top2 -> expert token lists ----------------

__global__ __launch_bounds__(256) void k_gate(const float* __restrict__ x,
                                              const float* __restrict__ gateW,
                                              int* __restrict__ counts,
                                              int* __restrict__ idx,
                                              float* __restrict__ wgt) {
    const int n = blockIdx.x, tid = threadIdx.x;
    float acc[NEXP] = {};
    for (int d = tid; d < DIM; d += 256) {
        float xv = x[(size_t)n * DIM + d];
        #pragma unroll
        for (int e = 0; e < NEXP; ++e) acc[e] += xv * gateW[(size_t)d * NEXP + e];
    }
    __shared__ float red[256];
    __shared__ float logits[NEXP];
    for (int e = 0; e < NEXP; ++e) {
        red[tid] = acc[e];
        __syncthreads();
        for (int s = 128; s > 0; s >>= 1) {
            if (tid < s) red[tid] += red[tid + s];
            __syncthreads();
        }
        if (tid == 0) logits[e] = red[0];
        __syncthreads();
    }
    if (tid == 0) {
        float m = logits[0];
        #pragma unroll
        for (int e = 1; e < NEXP; ++e) m = fmaxf(m, logits[e]);
        float p[NEXP];
        #pragma unroll
        for (int e = 0; e < NEXP; ++e) p[e] = __expf(logits[e] - m);
        int i0 = 0;
        #pragma unroll
        for (int e = 1; e < NEXP; ++e) if (p[e] > p[i0]) i0 = e;   // lowest index on ties
        int i1 = -1;
        #pragma unroll
        for (int e = 0; e < NEXP; ++e)
            if (e != i0 && (i1 < 0 || p[e] > p[i1])) i1 = e;
        float w0 = p[i0], w1 = p[i1], inv = 1.0f / (w0 + w1);
        w0 *= inv; w1 *= inv;
        int r0 = atomicAdd(&counts[i0], 1);
        idx[(size_t)i0 * NTOK + r0] = n; wgt[(size_t)i0 * NTOK + r0] = w0;
        int r1 = atomicAdd(&counts[i1], 1);
        idx[(size_t)i1 * NTOK + r1] = n; wgt[(size_t)i1 * NTOK + r1] = w1;
    }
}

// ---------------- WMMA GEMMs: block = 256 thr (8 waves), tile 128(M) x 64(N) ----------------
// Per wave: 16x64 tile, 1 A-frag + 4 B-frags + 4 v_wmma per K=32 step.

// Shared expert: d_out[n][c] = 0.5*gelu(x @ Ws + b)   (initializes d_out)
__global__ __launch_bounds__(256) void k_gemm_shared(const __bf16* __restrict__ xb,
                                                     const __bf16* __restrict__ wsT,
                                                     const float* __restrict__ sb,
                                                     float* __restrict__ out) {
    const int lane = threadIdx.x & 31, wave = threadIdx.x >> 5;
    const int nBase = blockIdx.x * 64, mBase = blockIdx.y * 128;
    const int kOff  = (lane & 16) ? 8 : 0;
    const int aRow  = mBase + wave * 16 + (lane & 15);
    const __bf16* aPtr = xb + (size_t)aRow * DIM + kOff;
    f32x8 acc[4] = {};
    for (int k0 = 0; k0 < DIM; k0 += 32) {
        bf16x16 a = load_frag(aPtr + k0);
        __builtin_prefetch(aPtr + k0 + 64, 0, 0);
        #pragma unroll
        for (int j = 0; j < 4; ++j) {
            const __bf16* bp = wsT + (size_t)(nBase + j * 16 + (lane & 15)) * DIM + kOff + k0;
            bf16x16 b = load_frag(bp);
            acc[j] = __builtin_amdgcn_wmma_f32_16x16x32_bf16(false, a, false, b, (short)0, acc[j], false, false);
        }
    }
    const int rBase = mBase + wave * 16 + ((lane & 16) ? 8 : 0);
    #pragma unroll
    for (int j = 0; j < 4; ++j) {
        int col = nBase + j * 16 + (lane & 15);
        float bias = sb[col];
        #pragma unroll
        for (int i = 0; i < 8; ++i)
            out[(size_t)(rBase + i) * DOUT + col] = 0.5f * gelu_exact(acc[j][i] + bias);
    }
}

// Expert GEMM1: h[r][c] = gelu(x[idx[r]] @ W1 + b1)   for r < cnt
__global__ __launch_bounds__(256) void k_gemm1(const __bf16* __restrict__ xb,
                                               const __bf16* __restrict__ w1T,
                                               const float* __restrict__ b1,
                                               const int* __restrict__ idxE,
                                               const int* __restrict__ cntE,
                                               __bf16* __restrict__ h) {
    const int cnt = cntE[0];
    const int mBase = blockIdx.y * 128;
    if (mBase >= cnt) return;                       // block-uniform: EXEC all-ones in WMMA loop
    const int lane = threadIdx.x & 31, wave = threadIdx.x >> 5;
    const int nBase = blockIdx.x * 64;
    const int kOff  = (lane & 16) ? 8 : 0;
    const int r     = mBase + wave * 16 + (lane & 15);
    int t = idxE[r];                                // in-bounds (capacity NTOK); value clamped below
    const int tok = (r < cnt) ? t : 0;
    const __bf16* aPtr = xb + (size_t)tok * DIM + kOff;
    f32x8 acc[4] = {};
    for (int k0 = 0; k0 < DIM; k0 += 32) {
        bf16x16 a = load_frag(aPtr + k0);
        #pragma unroll
        for (int j = 0; j < 4; ++j) {
            const __bf16* bp = w1T + (size_t)(nBase + j * 16 + (lane & 15)) * DIM + kOff + k0;
            bf16x16 b = load_frag(bp);
            acc[j] = __builtin_amdgcn_wmma_f32_16x16x32_bf16(false, a, false, b, (short)0, acc[j], false, false);
        }
    }
    const int rBase = mBase + wave * 16 + ((lane & 16) ? 8 : 0);
    #pragma unroll
    for (int j = 0; j < 4; ++j) {
        int col = nBase + j * 16 + (lane & 15);
        float bias = b1[col];
        #pragma unroll
        for (int i = 0; i < 8; ++i) {
            int rr = rBase + i;
            if (rr < cnt)
                h[(size_t)rr * HID + col] = (__bf16)gelu_exact(acc[j][i] + bias);
        }
    }
}

// Expert GEMM2: out[idx[r]][c] += w[r]*(h[r] @ W2 + b2)   for r < cnt
// (rows disjoint within an expert; experts serialized on the stream -> race-free)
__global__ __launch_bounds__(256) void k_gemm2(const __bf16* __restrict__ h,
                                               const __bf16* __restrict__ w2T,
                                               const float* __restrict__ b2,
                                               const int* __restrict__ idxE,
                                               const float* __restrict__ wgtE,
                                               const int* __restrict__ cntE,
                                               float* __restrict__ out) {
    const int cnt = cntE[0];
    const int mBase = blockIdx.y * 128;
    if (mBase >= cnt) return;
    const int lane = threadIdx.x & 31, wave = threadIdx.x >> 5;
    const int nBase = blockIdx.x * 64;
    const int kOff  = (lane & 16) ? 8 : 0;
    const int aRow  = mBase + wave * 16 + (lane & 15);   // arena row (garbage rows are finite, never stored)
    const __bf16* aPtr = h + (size_t)aRow * HID + kOff;
    f32x8 acc[4] = {};
    for (int k0 = 0; k0 < HID; k0 += 32) {
        bf16x16 a = load_frag(aPtr + k0);
        __builtin_prefetch(aPtr + k0 + 64, 0, 0);
        #pragma unroll
        for (int j = 0; j < 4; ++j) {
            const __bf16* bp = w2T + (size_t)(nBase + j * 16 + (lane & 15)) * HID + kOff + k0;
            bf16x16 b = load_frag(bp);
            acc[j] = __builtin_amdgcn_wmma_f32_16x16x32_bf16(false, a, false, b, (short)0, acc[j], false, false);
        }
    }
    const int rBase = mBase + wave * 16 + ((lane & 16) ? 8 : 0);
    #pragma unroll
    for (int j = 0; j < 4; ++j) {
        int col = nBase + j * 16 + (lane & 15);
        float bias = b2[col];
        #pragma unroll
        for (int i = 0; i < 8; ++i) {
            int rr = rBase + i;
            if (rr < cnt) {
                int tok = idxE[rr];
                float w = wgtE[rr];
                size_t o = (size_t)tok * DOUT + col;
                out[o] += w * (acc[j][i] + bias);
            }
        }
    }
}

// ---------------- LayerNorm in place on d_out ----------------

__global__ __launch_bounds__(256) void k_layernorm(float* __restrict__ out,
                                                   const float* __restrict__ gamma,
                                                   const float* __restrict__ beta) {
    const int n = blockIdx.x, tid = threadIdx.x;
    float* row = out + (size_t)n * DOUT;
    float loc[DOUT / 256];
    float s = 0.0f;
    #pragma unroll
    for (int i = 0; i < DOUT / 256; ++i) { loc[i] = row[tid + i * 256]; s += loc[i]; }
    __shared__ float red[256];
    red[tid] = s; __syncthreads();
    for (int st = 128; st > 0; st >>= 1) { if (tid < st) red[tid] += red[tid + st]; __syncthreads(); }
    float mean = red[0] * (1.0f / DOUT);
    __syncthreads();
    float s2 = 0.0f;
    #pragma unroll
    for (int i = 0; i < DOUT / 256; ++i) { float d = loc[i] - mean; s2 += d * d; }
    red[tid] = s2; __syncthreads();
    for (int st = 128; st > 0; st >>= 1) { if (tid < st) red[tid] += red[tid + st]; __syncthreads(); }
    float rstd = rsqrtf(red[0] * (1.0f / DOUT) + LN_EPS);
    __syncthreads();
    #pragma unroll
    for (int i = 0; i < DOUT / 256; ++i) {
        int c = tid + i * 256;
        row[c] = (loc[i] - mean) * rstd * gamma[c] + beta[c];
    }
}

// ---------------- host launcher ----------------

extern "C" void kernel_launch(void* const* d_in, const int* in_sizes, int n_in,
                              void* d_out, int out_size, void* d_ws, size_t ws_size,
                              hipStream_t stream) {
    const float* x        = (const float*)d_in[0];
    const float* W1       = (const float*)d_in[1];
    const float* b1       = (const float*)d_in[2];
    const float* W2       = (const float*)d_in[3];
    const float* b2       = (const float*)d_in[4];
    const float* gate_W   = (const float*)d_in[5];
    const float* shared_W = (const float*)d_in[6];
    const float* shared_b = (const float*)d_in[7];
    const float* gamma    = (const float*)d_in[8];
    const float* beta     = (const float*)d_in[9];
    float* out = (float*)d_out;

    char* w = (char*)d_ws;
    __bf16* xb  = (__bf16*)(w + OFF_XB);
    __bf16* w1T = (__bf16*)(w + OFF_W1T);
    __bf16* w2T = (__bf16*)(w + OFF_W2T);
    __bf16* wsT = (__bf16*)(w + OFF_WST);
    __bf16* hA  = (__bf16*)(w + OFF_H);
    int*    idx = (int*)  (w + OFF_IDX);
    float*  wgt = (float*)(w + OFF_WGT);
    int*    cnt = (int*)  (w + OFF_CNT);

    // prep
    k_init_counts<<<1, 32, 0, stream>>>(cnt);
    k_convert_x<<<(NTOK * DIM / 4 + 255) / 256, 256, 0, stream>>>(x, xb);
    k_transpose_bf16<<<dim3(HID / 32, DIM / 32, NEXP), dim3(32, 8), 0, stream>>>(W1, w1T, DIM, HID);
    k_transpose_bf16<<<dim3(DOUT / 32, HID / 32, NEXP), dim3(32, 8), 0, stream>>>(W2, w2T, HID, DOUT);
    k_transpose_bf16<<<dim3(DOUT / 32, DIM / 32, 1), dim3(32, 8), 0, stream>>>(shared_W, wsT, DIM, DOUT);

    // routing
    k_gate<<<NTOK, 256, 0, stream>>>(x, gate_W, cnt, idx, wgt);

    // shared expert (initializes d_out)
    k_gemm_shared<<<dim3(DOUT / 64, NTOK / 128), 256, 0, stream>>>(xb, wsT, shared_b, out);

    // routed experts, serialized on the stream
    for (int e = 0; e < NEXP; ++e) {
        k_gemm1<<<dim3(HID / 64, NTOK / 128), 256, 0, stream>>>(
            xb, w1T + (size_t)e * HID * DIM, b1 + (size_t)e * HID,
            idx + (size_t)e * NTOK, cnt + e, hA);
        k_gemm2<<<dim3(DOUT / 64, NTOK / 128), 256, 0, stream>>>(
            hA, w2T + (size_t)e * DOUT * HID, b2 + (size_t)e * DOUT,
            idx + (size_t)e * NTOK, wgt + (size_t)e * NTOK, cnt + e, out);
    }

    // layernorm in place
    k_layernorm<<<NTOK, 256, 0, stream>>>(out, gamma, beta);
}
